// RPNModule_730144440350
// MI455X (gfx1250) — matrix operand order repaired and checked by reading, not compiled, not run on hardware
//
#include <hip/hip_runtime.h>
#include <math.h>
#include <stdint.h>

typedef __attribute__((ext_vector_type(16))) _Float16 v16h;
typedef __attribute__((ext_vector_type(8)))  _Float16 v8h;
typedef __attribute__((ext_vector_type(8)))  float    v8f;

#define B_      4
#define H_      200
#define W_      176
#define HW_     (H_*W_)      /* 35200 */
#define CIN_    256
#define CM_     64
#define NC_     3
#define K_      500
#define PRED_C  11

// ---------------------------------------------------------------------------
// bev_feats f32 [CIN][HW] (one batch) -> f16 [HW][CIN]
__global__ void pack_bev_kernel(const float* __restrict__ src, _Float16* __restrict__ dst) {
    int t = blockIdx.x * 256 + threadIdx.x;           // over HW_*CIN_
    if (t >= HW_ * CIN_) return;
    int p = t / CIN_, c = t % CIN_;
    dst[t] = (_Float16)src[c * HW_ + p];
}

// weights OIHW f32 [O][I][3][3] -> f16 [O][9*I] with k = tap*I + ic
__global__ void repack_w_kernel(const float* __restrict__ w, _Float16* __restrict__ wf,
                                int O, int I) {
    int t = blockIdx.x * 256 + threadIdx.x;
    if (t >= O * I * 9) return;
    int o = t / (I * 9);
    int ic = (t / 9) % I;
    int tap = t % 9;
    wf[o * (9 * I) + tap * I + ic] = (_Float16)w[t];
}

// ---------------------------------------------------------------------------
// Implicit-GEMM 3x3 SAME conv, CIN in {256,64} (compile-time), Cout=64,
// fused BN + ReLU.
// in  : f16 [HW][CIN] (HWC), per-batch via blockIdx.y*inStride
// wf  : f16 [64][9*CIN]   (k = tap*CIN + ic)
// out : f16 [HW][64]
// Block = 256 threads (8 waves). Tile: 64 oc x 128 px. Wave: 16 oc x 64 px
// (four 16x16 f32 accumulators; A fragment reused across the 4 WMMAs of each
// 32-wide K-step). K-step = one 3x3 tap x 32 ic, contiguous in HWC layout, so
// every B fragment is exactly two global_load_b128.
// Per-tap weight slab (64 oc x CIN) is double-buffered in LDS via
// GLOBAL_LOAD_ASYNC_TO_LDS_B128; A fragments come from ds_load_b128.
template <int CIN>
__global__ __launch_bounds__(256) void conv_wmma_kernel(
    const _Float16* __restrict__ in0, int inStride,
    const _Float16* __restrict__ wf,
    const float* __restrict__ bias, const float* __restrict__ g,
    const float* __restrict__ be,   const float* __restrict__ mu,
    const float* __restrict__ var,
    _Float16* __restrict__ out0, int outStride)
{
#if defined(__HIP_DEVICE_COMPILE__)
    constexpr int NT = 4;                          // four 16-px N-tiles / wave
    __shared__ _Float16 asl[2][CM_ * CIN];         // double-buffered A slab
    const _Float16* in  = in0  + (size_t)blockIdx.y * inStride;
    _Float16*       out = out0 + (size_t)blockIdx.y * outStride;
    const int lane   = threadIdx.x & 31;
    const int wave   = threadIdx.x >> 5;
    const int ocBase = (wave & 3) << 4;
    const int pxBase = blockIdx.x * 128 + ((wave >> 2) << 6);
    const int mg = lane >> 4;                      // lane group (0/1)
    const int ln = lane & 15;

    // cooperative async stage of one tap's weight slab into LDS buffer `buf`
    auto stage = [&](int tap, int buf) {
        constexpr int CPS = CIN / 8;               // 16B chunks per oc row
        const _Float16* src = wf + tap * CIN;
        #pragma unroll
        for (int i = 0; i < CIN / 32; ++i) {       // 8*CIN chunks / 256 thr
            int c   = threadIdx.x + i * 256;
            int oc  = c / CPS;
            int col = (c % CPS) * 8;
            const _Float16* gp = src + oc * 9 * CIN + col;
            uint32_t ldsOff = (uint32_t)(uintptr_t)&asl[buf][oc * CIN + col];
            asm volatile("global_load_async_to_lds_b128 %0, %1, off"
                         :: "v"(ldsOff), "v"(gp) : "memory");
        }
    };

    int p[NT], pyv[NT], pxv[NT];
    #pragma unroll
    for (int n = 0; n < NT; ++n) {
        p[n]   = pxBase + 16 * n + ln;
        pyv[n] = p[n] / W_;
        pxv[n] = p[n] % W_;
    }

    v8f zero = {};
    v8f acc[NT];
    #pragma unroll
    for (int n = 0; n < NT; ++n) acc[n] = zero;

    stage(0, 0);
    for (int tap = 0; tap < 9; ++tap) {
        const int buf = tap & 1;
        asm volatile("s_wait_asynccnt 0" ::: "memory");
        __syncthreads();                           // slab ready; prev readers done
        if (tap < 8) stage(tap + 1, buf ^ 1);      // overlap next tap's DMA

        const int dy = tap / 3 - 1, dx = tap % 3 - 1;
        const _Float16* bp[NT];
        bool vld[NT];
        #pragma unroll
        for (int n = 0; n < NT; ++n) {
            int yy = pyv[n] + dy, xx = pxv[n] + dx;
            vld[n] = ((unsigned)yy < (unsigned)H_) && ((unsigned)xx < (unsigned)W_);
            bp[n] = in + (yy * W_ + xx) * CIN + mg * 8;
        }
        // A fragment rows: oc = ocBase+ln; K chunks at +mg*8 and +16+mg*8
        const int arow = (ocBase + ln) * CIN + mg * 8;
        #pragma unroll
        for (int kb = 0; kb < CIN; kb += 32) {
            v8h alo = *(const v8h*)&asl[buf][arow + kb];
            v8h ahi = *(const v8h*)&asl[buf][arow + kb + 16];
            v16h A = __builtin_shufflevector(alo, ahi,
                0,1,2,3,4,5,6,7,8,9,10,11,12,13,14,15);
            #pragma unroll
            for (int n = 0; n < NT; ++n) {
                v8h z = {};
                v8h bl = vld[n] ? *(const v8h*)(bp[n] + kb)     : z;
                v8h bh = vld[n] ? *(const v8h*)(bp[n] + kb + 16) : z;
                v16h Bm = __builtin_shufflevector(bl, bh,
                    0,1,2,3,4,5,6,7,8,9,10,11,12,13,14,15);
                acc[n] = __builtin_amdgcn_wmma_f32_16x16x32_f16(
                    false, A, false, Bm, (short)0, acc[n], false, false);
            }
        }
    }
    // Epilogue: C/D layout — lane group 0: M=r, group 1: M=r+8; N=lane%16
    #pragma unroll
    for (int r = 0; r < 8; ++r) {
        int oc = ocBase + r + mg * 8;
        float s  = g[oc] * rsqrtf(var[oc] + 1e-5f);
        float sh = be[oc] + (bias[oc] - mu[oc]) * s;
        #pragma unroll
        for (int n = 0; n < NT; ++n)
            out[p[n] * CM_ + oc] = (_Float16)fmaxf(acc[n][r] * s + sh, 0.0f);
    }
#else
    (void)in0;(void)inStride;(void)wf;(void)bias;(void)g;(void)be;(void)mu;(void)var;
    (void)out0;(void)outStride;
#endif
}

// ---------------------------------------------------------------------------
// Final 3x3 convs (64 -> few channels), f32 OIHW weights, no BN/ReLU.
// h: f16 [B][HW][64]; preds: f32 [B][11][HW]
__global__ void conv_final_kernel(const _Float16* __restrict__ h,
                                  const float* __restrict__ w,
                                  const float* __restrict__ bias,
                                  float* __restrict__ preds, int ocOff)
{
    int p = blockIdx.x * 256 + threadIdx.x;
    if (p >= HW_) return;
    int oc = blockIdx.y, b = blockIdx.z;
    const _Float16* hb = h + (size_t)b * HW_ * CM_;
    int y = p / W_, x = p % W_;
    float acc = bias[oc];
    for (int tap = 0; tap < 9; ++tap) {
        int dy = tap / 3 - 1, dx = tap % 3 - 1;
        int yy = y + dy, xx = x + dx;
        if ((unsigned)yy >= (unsigned)H_ || (unsigned)xx >= (unsigned)W_) continue;
        const _Float16* hp = hb + (yy * W_ + xx) * CM_;
        const float* wp = w + oc * CM_ * 9 + tap;
        #pragma unroll 8
        for (int ic = 0; ic < CM_; ++ic)
            acc += (float)hp[ic] * wp[ic * 9];
    }
    preds[((size_t)b * PRED_C + ocOff + oc) * HW_ + p] = acc;
}

// sigmoid over hm channels (8..10) -> scores [B][3][HW]
__global__ void sigmoid_kernel(const float* __restrict__ preds, float* __restrict__ scores) {
    int t = blockIdx.x * 256 + threadIdx.x;
    if (t >= B_ * NC_ * HW_) return;
    int b = t / (NC_ * HW_), r = t % (NC_ * HW_);
    float v = preds[((size_t)b * PRED_C + 8) * HW_ + r];
    scores[t] = 1.0f / (1.0f + expf(-v));
}

// per-(b,class) top-K over HW via iterative argmax (destructive on scores)
__global__ __launch_bounds__(256) void topk_class_kernel(float* __restrict__ scores,
                                                         float* __restrict__ s1,
                                                         int* __restrict__ i1) {
    int bc = blockIdx.x, tid = threadIdx.x;
    float* sc = scores + (size_t)bc * HW_;
    __shared__ float bv[256]; __shared__ int bi[256];
    for (int it = 0; it < K_; ++it) {
        float best = -1e30f; int bidx = HW_;
        for (int p = tid; p < HW_; p += 256) {
            float v = sc[p];
            if (v > best || (v == best && p < bidx)) { best = v; bidx = p; }
        }
        bv[tid] = best; bi[tid] = bidx; __syncthreads();
        for (int s = 128; s > 0; s >>= 1) {
            if (tid < s) {
                if (bv[tid+s] > bv[tid] || (bv[tid+s] == bv[tid] && bi[tid+s] < bi[tid])) {
                    bv[tid] = bv[tid+s]; bi[tid] = bi[tid+s];
                }
            }
            __syncthreads();
        }
        if (tid == 0) { s1[bc*K_+it] = bv[0]; i1[bc*K_+it] = bi[0]; sc[bi[0]] = -1e30f; }
        __syncthreads();
    }
}

// per-batch global top-K over NC*K candidates
__global__ __launch_bounds__(256) void topk_global_kernel(const float* __restrict__ s1,
                                                          const int* __restrict__ i1,
                                                          float* __restrict__ s2,
                                                          int* __restrict__ cls,
                                                          int* __restrict__ inds) {
    int b = blockIdx.x, tid = threadIdx.x;
    __shared__ float ls[NC_*K_];
    __shared__ float bv[256]; __shared__ int bi[256];
    for (int t = tid; t < NC_*K_; t += 256) ls[t] = s1[b*NC_*K_ + t];
    __syncthreads();
    for (int it = 0; it < K_; ++it) {
        float best = -1e30f; int bidx = NC_*K_;
        for (int t = tid; t < NC_*K_; t += 256) {
            float v = ls[t];
            if (v > best || (v == best && t < bidx)) { best = v; bidx = t; }
        }
        bv[tid] = best; bi[tid] = bidx; __syncthreads();
        for (int s = 128; s > 0; s >>= 1) {
            if (tid < s) {
                if (bv[tid+s] > bv[tid] || (bv[tid+s] == bv[tid] && bi[tid+s] < bi[tid])) {
                    bv[tid] = bv[tid+s]; bi[tid] = bi[tid+s];
                }
            }
            __syncthreads();
        }
        if (tid == 0) {
            int idx = bi[0];
            s2[b*K_+it]  = bv[0];
            cls[b*K_+it] = idx / K_;
            inds[b*K_+it] = i1[b*NC_*K_ + idx];
            ls[idx] = -1e30f;
        }
        __syncthreads();
    }
}

// gather + box decode (edge-form + BEV center boxes + validity)
__global__ void decode_kernel(const float* __restrict__ preds, const float* __restrict__ s2,
                              const int* __restrict__ inds,
                              float* __restrict__ boxes, float* __restrict__ bev,
                              int* __restrict__ valid) {
    int t = blockIdx.x * 256 + threadIdx.x;
    if (t >= B_ * K_) return;
    int b = t / K_;
    int idx = inds[t];
    float ysf = (float)(idx / W_), xsf = (float)(idx % W_);
    const float* pb = preds + (size_t)b * PRED_C * HW_;
    float cx = pb[0*HW_+idx], cy = pb[1*HW_+idx];
    float z  = pb[2*HW_+idx];
    float d0 = expf(pb[3*HW_+idx]), d1 = expf(pb[4*HW_+idx]), d2 = expf(pb[5*HW_+idx]);
    float ang = atan2f(pb[7*HW_+idx], pb[6*HW_+idx]);
    float xw = (xsf + cx) * 0.4f + 0.0f;      // STRIDE*VX + PCR_X
    float yw = (ysf + cy) * 0.4f - 40.0f;     // STRIDE*VY + PCR_Y
    float sc = s2[t];
    int va = (sc > 0.1f)
        && xw >= -10.f && xw <= 80.f
        && yw >= -50.f && yw <= 50.f
        && z  >= -10.f && z  <= 10.f;
    float* bx = boxes + t * 7;
    bx[0] = xw - 0.5f*d0; bx[1] = yw - 0.5f*d1; bx[2] = z - 0.5f*d2;
    bx[3] = d0; bx[4] = d1; bx[5] = d2; bx[6] = ang;
    float* bv = bev + t * 5;
    bv[0] = xw; bv[1] = yw; bv[2] = d0; bv[3] = d1; bv[4] = ang;
    valid[t] = va;
}

// stable descending sort by masked score via rank counting (O(K^2))
__global__ __launch_bounds__(512) void order_kernel(
    const float* __restrict__ boxes, const float* __restrict__ bev,
    const float* __restrict__ s2, const int* __restrict__ cls,
    const int* __restrict__ valid,
    float* __restrict__ obox, float* __restrict__ obev,
    float* __restrict__ osc, int* __restrict__ ocl, int* __restrict__ ova)
{
    int b = blockIdx.x, i = threadIdx.x;
    __shared__ float key[K_];
    if (i < K_) key[i] = valid[b*K_+i] ? s2[b*K_+i] : -1e30f;
    __syncthreads();
    if (i < K_) {
        float ki = key[i]; int r = 0;
        for (int j = 0; j < K_; ++j) {
            float kj = key[j];
            r += (kj > ki) || (kj == ki && j < i);
        }
        int src = b*K_ + i, dst = b*K_ + r;
        for (int q = 0; q < 7; ++q) obox[dst*7+q] = boxes[src*7+q];
        for (int q = 0; q < 5; ++q) obev[dst*5+q] = bev[src*5+q];
        osc[dst] = s2[src]; ocl[dst] = cls[src]; ova[dst] = valid[src];
    }
}

// rotated-rect pairwise IoU via Sutherland-Hodgman (CAP=8)
__global__ void iou_kernel(const float* __restrict__ obev, float* __restrict__ iou) {
    int t = blockIdx.x * 256 + threadIdx.x;
    if (t >= B_ * K_ * K_) return;
    int b = t / (K_*K_), r = t % (K_*K_);
    int i = r / K_, j = r % K_;
    const float* A  = obev + (size_t)(b*K_ + i) * 5;
    const float* Bb = obev + (size_t)(b*K_ + j) * 5;
    const float ddx[4] = {-0.5f, 0.5f, 0.5f, -0.5f};
    const float ddy[4] = {-0.5f,-0.5f, 0.5f,  0.5f};
    float qx[4], qy[4];
    {
        float c = cosf(Bb[4]), s = sinf(Bb[4]);
        for (int k = 0; k < 4; ++k) {
            float dx = ddx[k]*Bb[2], dy = ddy[k]*Bb[3];
            qx[k] = Bb[0] + dx*c - dy*s;
            qy[k] = Bb[1] + dx*s + dy*c;
        }
    }
    float px[8], py[8]; int cnt = 4;
    {
        float c = cosf(A[4]), s = sinf(A[4]);
        for (int k = 0; k < 4; ++k) {
            float dx = ddx[k]*A[2], dy = ddy[k]*A[3];
            px[k] = A[0] + dx*c - dy*s;
            py[k] = A[1] + dx*s + dy*c;
        }
        for (int k = 4; k < 8; ++k) { px[k] = 0.f; py[k] = 0.f; }
    }
    for (int e = 0; e < 4; ++e) {
        float ax = qx[e], ay = qy[e];
        float ex = qx[(e+1)&3] - ax, ey = qy[(e+1)&3] - ay;
        float nx[8], ny[8]; int m = 0;
        for (int v = 0; v < cnt; ++v) {
            int w2 = (v + 1 >= cnt) ? 0 : v + 1;
            float c0x = px[v],  c0y = py[v];
            float c1x = px[w2], c1y = py[w2];
            float d0 = ex*(c0y-ay) - ey*(c0x-ax);
            float d1 = ex*(c1y-ay) - ey*(c1x-ax);
            bool in0 = d0 >= 0.f, in1 = d1 >= 0.f;
            if (in0 && m < 8) { nx[m] = c0x; ny[m] = c0y; ++m; }
            if ((in0 != in1) && m < 8) {
                float den = d0 - d1;
                float tt = d0 / (fabsf(den) < 1e-12f ? 1e-12f : den);
                nx[m] = c0x + tt*(c1x-c0x); ny[m] = c0y + tt*(c1y-c0y); ++m;
            }
        }
        cnt = m;
        for (int v = 0; v < m; ++v) { px[v] = nx[v]; py[v] = ny[v]; }
    }
    float area = 0.f;
    for (int v = 0; v < cnt; ++v) {
        int w2 = (v + 1 >= cnt) ? 0 : v + 1;
        area += px[v]*py[w2] - py[v]*px[w2];
    }
    float inter = 0.5f * fabsf(area);
    float uni = A[2]*A[3] + Bb[2]*Bb[3] - inter;
    iou[t] = inter / fmaxf(uni, 1e-8f);
}

// sequential greedy suppression (per-class, iou > 0.01)
__global__ __launch_bounds__(512) void nms_kernel(const float* __restrict__ iou,
                                                  const int* __restrict__ ocl,
                                                  const int* __restrict__ ova,
                                                  int* __restrict__ keep) {
    int b = blockIdx.x, tid = threadIdx.x;
    __shared__ int kp[K_]; __shared__ int cl[K_];
    if (tid < K_) { kp[tid] = ova[b*K_+tid]; cl[tid] = ocl[b*K_+tid]; }
    __syncthreads();
    for (int i = 0; i < K_; ++i) {
        int ki = kp[i], ci = cl[i];
        if (ki) {
            for (int j = tid; j < K_; j += 512) {
                if (j > i && cl[j] == ci && iou[(size_t)(b*K_+i)*K_+j] > 0.01f) kp[j] = 0;
            }
        }
        __syncthreads();
    }
    if (tid < K_) keep[b*K_+tid] = kp[tid];
}

// class-keyed stable output permutation + write rois/scores/labels
__global__ __launch_bounds__(512) void writeout_kernel(const float* __restrict__ obox,
                                                       const float* __restrict__ osc,
                                                       const int* __restrict__ ocl,
                                                       const int* __restrict__ keep,
                                                       float* __restrict__ out) {
    int b = blockIdx.x, i = threadIdx.x;
    __shared__ int key[K_];
    if (i < K_) key[i] = keep[b*K_+i] ? ocl[b*K_+i]*(K_+1) + i
                                      : NC_*(K_+1) + K_ + i;
    __syncthreads();
    if (i < K_) {
        int ki = key[i], r = 0;
        for (int j = 0; j < K_; ++j) r += (key[j] < ki);
        int kp = keep[b*K_+i];
        float* rois = out;
        float* scs  = out + B_*K_*7;
        float* lbs  = out + B_*K_*8;
        int dst = b*K_ + r, src = b*K_ + i;
        for (int q = 0; q < 7; ++q) rois[dst*7+q] = kp ? obox[src*7+q] : 0.f;
        scs[dst] = kp ? osc[src] : 0.f;
        lbs[dst] = kp ? (float)ocl[src] : 0.f;
    }
}

// ---------------------------------------------------------------------------
static inline size_t al256(size_t x) { return (x + 255) & ~(size_t)255; }

extern "C" void kernel_launch(void* const* d_in, const int* in_sizes, int n_in,
                              void* d_out, int out_size, void* d_ws, size_t ws_size,
                              hipStream_t stream) {
    (void)in_sizes; (void)n_in; (void)out_size; (void)ws_size;
    const float* bev_feats = (const float*)d_in[0];
    const float* sw  = (const float*)d_in[1];
    const float* sb  = (const float*)d_in[2];
    const float* s_g = (const float*)d_in[3];
    const float* s_b = (const float*)d_in[4];
    const float* s_m = (const float*)d_in[5];
    const float* s_v = (const float*)d_in[6];
    const float* hw1 = (const float*)d_in[7];
    const float* hb1 = (const float*)d_in[8];
    const float* h_g = (const float*)d_in[9];
    const float* h_b = (const float*)d_in[10];
    const float* h_m = (const float*)d_in[11];
    const float* h_v = (const float*)d_in[12];
    const float* wfin[5]  = { (const float*)d_in[13], (const float*)d_in[15],
                              (const float*)d_in[17], (const float*)d_in[19],
                              (const float*)d_in[21] };
    const float* bfin[5]  = { (const float*)d_in[14], (const float*)d_in[16],
                              (const float*)d_in[18], (const float*)d_in[20],
                              (const float*)d_in[22] };
    const int ocFin[5]  = { 2, 1, 3, 2, 3 };
    const int ocOff[5]  = { 0, 2, 3, 6, 8 };
    float* out = (float*)d_out;

    // -------- workspace carve --------
    uint8_t* w = (uint8_t*)d_ws;
    size_t off = 0;
    #define CARVE(name, bytes) uint8_t* name = w + off; off = al256(off + (bytes))
    CARVE(ws_bevh,  (size_t)HW_*CIN_*2);          // one batch, f16 HWC
    CARVE(ws_x,     (size_t)B_*HW_*CM_*2);        // shared-conv out, f16 HWC
    CARVE(ws_h,     (size_t)B_*HW_*CM_*2);        // head mid out (reused), f16
    CARVE(ws_preds, (size_t)B_*PRED_C*HW_*4);     // f32 CHW
    CARVE(ws_score, (size_t)B_*NC_*HW_*4);
    CARVE(ws_wfs,   (size_t)CM_*9*CIN_*2);
    CARVE(ws_wfm,   (size_t)5*CM_*9*CM_*2);
    CARVE(ws_s1,    (size_t)B_*NC_*K_*4);
    CARVE(ws_i1,    (size_t)B_*NC_*K_*4);
    CARVE(ws_s2,    (size_t)B_*K_*4);
    CARVE(ws_cls,   (size_t)B_*K_*4);
    CARVE(ws_inds,  (size_t)B_*K_*4);
    CARVE(ws_box,   (size_t)B_*K_*7*4);
    CARVE(ws_bev5,  (size_t)B_*K_*5*4);
    CARVE(ws_val,   (size_t)B_*K_*4);
    CARVE(ws_obox,  (size_t)B_*K_*7*4);
    CARVE(ws_obev,  (size_t)B_*K_*5*4);
    CARVE(ws_osc,   (size_t)B_*K_*4);
    CARVE(ws_ocl,   (size_t)B_*K_*4);
    CARVE(ws_ova,   (size_t)B_*K_*4);
    CARVE(ws_iou,   (size_t)B_*K_*K_*4);
    CARVE(ws_keep,  (size_t)B_*K_*4);
    #undef CARVE

    _Float16* bevh = (_Float16*)ws_bevh;
    _Float16* xbuf = (_Float16*)ws_x;
    _Float16* hbuf = (_Float16*)ws_h;
    float*    preds = (float*)ws_preds;
    float*    scores = (float*)ws_score;
    _Float16* wfs = (_Float16*)ws_wfs;
    _Float16* wfm = (_Float16*)ws_wfm;

    // -------- weight repack --------
    repack_w_kernel<<<(CM_*CIN_*9 + 255)/256, 256, 0, stream>>>(sw, wfs, CM_, CIN_);
    for (int i = 0; i < 5; ++i)
        repack_w_kernel<<<(CM_*CM_*9 + 255)/256, 256, 0, stream>>>(
            hw1 + (size_t)i*CM_*CM_*9, wfm + (size_t)i*CM_*9*CM_, CM_, CM_);

    // -------- shared conv (per-batch to reuse the f16 staging buffer) --------
    for (int b = 0; b < B_; ++b) {
        pack_bev_kernel<<<(HW_*CIN_ + 255)/256, 256, 0, stream>>>(
            bev_feats + (size_t)b*CIN_*HW_, bevh);
        conv_wmma_kernel<CIN_><<<dim3(HW_/128, 1), 256, 0, stream>>>(
            bevh, 0, wfs, sb, s_g, s_b, s_m, s_v,
            xbuf + (size_t)b*HW_*CM_, 0);
    }

    // -------- heads: mid conv (WMMA) + final conv --------
    for (int i = 0; i < 5; ++i) {
        conv_wmma_kernel<CM_><<<dim3(HW_/128, B_), 256, 0, stream>>>(
            xbuf, HW_*CM_, wfm + (size_t)i*CM_*9*CM_,
            hb1 + (size_t)i*CM_, h_g + (size_t)i*CM_, h_b + (size_t)i*CM_,
            h_m + (size_t)i*CM_, h_v + (size_t)i*CM_,
            hbuf, HW_*CM_);
        conv_final_kernel<<<dim3((HW_+255)/256, ocFin[i], B_), 256, 0, stream>>>(
            hbuf, wfin[i], bfin[i], preds, ocOff[i]);
    }

    // -------- post-processing --------
    sigmoid_kernel<<<(B_*NC_*HW_ + 255)/256, 256, 0, stream>>>(preds, scores);
    topk_class_kernel<<<B_*NC_, 256, 0, stream>>>(scores, (float*)ws_s1, (int*)ws_i1);
    topk_global_kernel<<<B_, 256, 0, stream>>>((float*)ws_s1, (int*)ws_i1,
        (float*)ws_s2, (int*)ws_cls, (int*)ws_inds);
    decode_kernel<<<(B_*K_ + 255)/256, 256, 0, stream>>>(preds, (float*)ws_s2,
        (int*)ws_inds, (float*)ws_box, (float*)ws_bev5, (int*)ws_val);
    order_kernel<<<B_, 512, 0, stream>>>((float*)ws_box, (float*)ws_bev5,
        (float*)ws_s2, (int*)ws_cls, (int*)ws_val,
        (float*)ws_obox, (float*)ws_obev, (float*)ws_osc, (int*)ws_ocl, (int*)ws_ova);
    iou_kernel<<<(B_*K_*K_ + 255)/256, 256, 0, stream>>>((float*)ws_obev, (float*)ws_iou);
    nms_kernel<<<B_, 512, 0, stream>>>((float*)ws_iou, (int*)ws_ocl, (int*)ws_ova,
        (int*)ws_keep);
    writeout_kernel<<<B_, 512, 0, stream>>>((float*)ws_obox, (float*)ws_osc,
        (int*)ws_ocl, (int*)ws_keep, out);
}
